// SIRENAutodecoder_mdf_film_32418413150758
// MI455X (gfx1250) — compile-verified, matrix operand order checked by reading, not run on hardware
//
#include <hip/hip_runtime.h>

#define NT     32            // frames
#define HPIX   4096          // 64*64 pixels per frame
#define CIN    3
#define LAT    256
#define HID    256
#define NOUT   3
#define NHL    3
#define OMEGA  30.0f

#define TILE_P 128           // pixels per block in siren kernel
#define PITCH  264           // LDS row pitch in halves (528B = 132 dwords -> conflict-free b128)

typedef __attribute__((ext_vector_type(16))) _Float16 v16h;
typedef __attribute__((ext_vector_type(8)))  _Float16 v8h;
typedef __attribute__((ext_vector_type(8)))  float    v8f;
typedef __attribute__((ext_vector_type(4)))  float    v4f;

union U16H { v16h v; v8h h[2]; };

// Fragment-packed weight layout for one (layer, frame) 256x256 matrix:
//   fragment f = n_tile*8 + k_tile   (n_tile = out/16, k_tile = in/32), 512 halves each
//   within fragment: lane L owns 16 contiguous halves at f*512 + L*16
//   lane L -> out col n = n_tile*16 + (L&15); halves j<8 -> k = k0 + 8*(L>>4) + j,
//                                             halves j>=8 -> k = k0 + 8*(L>>4) + 16 + (j-8)
// This matches the V_WMMA 16-bit B operand layout exactly, and makes each wave's
// B-fragment load one contiguous 1KB block (perfect L2 coalescing).

// ---------------------------------------------------------------------------
// Kernel 1a: per-frame effective hidden weights (fp16, fragment-packed)
//   Weffh[i][t] = Wh[i] + latents[t] @ HWh[i].T     (memory bound on HWh, 201MB)
// ---------------------------------------------------------------------------
__global__ __launch_bounds__(256)
void hyper_weights_kernel(const float* __restrict__ latents,
                          const float* __restrict__ Wh,
                          const float* __restrict__ HWh,
                          _Float16* __restrict__ weffh) {
  __shared__ float lat[NT * LAT];                       // 32 KB
  for (int i = threadIdx.x; i < NT * LAT; i += 256) lat[i] = latents[i];
  __syncthreads();

  const long ng = (long)blockIdx.x * 256 + threadIdx.x; // 0 .. 3*65536-1
  const float* __restrict__ row = HWh + ng * LAT;

  float acc[NT];
#pragma unroll
  for (int t = 0; t < NT; ++t) acc[t] = 0.0f;

  for (int k = 0; k < LAT; k += 4) {
    v4f w = *(const v4f*)(row + k);
#pragma unroll
    for (int t = 0; t < NT; ++t) {
      const float* l = lat + t * LAT + k;
      acc[t] = fmaf(w.x, l[0], fmaf(w.y, l[1], fmaf(w.z, l[2], fmaf(w.w, l[3], acc[t]))));
    }
  }

  const float base = Wh[ng];
  const int n   = (int)(ng & 65535);                    // (o, jin) within layer
  const int i   = (int)(ng >> 16);                      // layer
  const int o   = n >> 8;
  const int jin = n & 255;
  // fragment-packed destination offset (bijective 0..65535)
  const int ktile = jin >> 5;
  const int kk    = jin & 31;
  const int lh    = (kk >> 3) & 1;
  const int jj    = (kk & 7) | ((kk >> 4) << 3);
  const int lanep = (o & 15) | (lh << 4);
  const long frag = ((long)((o >> 4) * 8 + ktile) << 9) + (lanep << 4) + jj;
#pragma unroll
  for (int t = 0; t < NT; ++t)
    weffh[(((long)(i * NT + t)) << 16) + frag] = (_Float16)(acc[t] + base);
}

// ---------------------------------------------------------------------------
// Kernel 1b: layer-0 effective weights (fp32, tiny): Weff0[t][o][i]
// ---------------------------------------------------------------------------
__global__ __launch_bounds__(256)
void hyper_w0_kernel(const float* __restrict__ latents,
                     const float* __restrict__ W0,
                     const float* __restrict__ HW0,
                     float* __restrict__ weff0) {
  __shared__ float lat[NT * LAT];
  for (int i = threadIdx.x; i < NT * LAT; i += 256) lat[i] = latents[i];
  __syncthreads();

  const int n = blockIdx.x * 256 + threadIdx.x;         // 0..767
  const float* __restrict__ row = HW0 + (long)n * LAT;

  float acc[NT];
#pragma unroll
  for (int t = 0; t < NT; ++t) acc[t] = 0.0f;
  for (int k = 0; k < LAT; k += 4) {
    v4f w = *(const v4f*)(row + k);
#pragma unroll
    for (int t = 0; t < NT; ++t) {
      const float* l = lat + t * LAT + k;
      acc[t] = fmaf(w.x, l[0], fmaf(w.y, l[1], fmaf(w.z, l[2], fmaf(w.w, l[3], acc[t]))));
    }
  }
  const float base = W0[n];
#pragma unroll
  for (int t = 0; t < NT; ++t)
    weff0[t * (HID * CIN) + n] = acc[t] + base;
}

// ---------------------------------------------------------------------------
// Kernel 1c: effective biases beff[t][k][h], k=0..NHL (tiny)
// ---------------------------------------------------------------------------
__global__ __launch_bounds__(256)
void hyper_bias_kernel(const float* __restrict__ latents,
                       const float* __restrict__ b0,
                       const float* __restrict__ bh,
                       const float* __restrict__ HB,
                       float* __restrict__ beff) {
  __shared__ float lat[NT * LAT];
  for (int i = threadIdx.x; i < NT * LAT; i += 256) lat[i] = latents[i];
  __syncthreads();

  const int n  = blockIdx.x * 256 + threadIdx.x;        // 0..1023 = (k,h)
  const int kk = n >> 8;
  const int h  = n & 255;
  const float* __restrict__ row = HB + (long)n * LAT;

  float acc[NT];
#pragma unroll
  for (int t = 0; t < NT; ++t) acc[t] = 0.0f;
  for (int k = 0; k < LAT; k += 4) {
    v4f w = *(const v4f*)(row + k);
#pragma unroll
    for (int t = 0; t < NT; ++t) {
      const float* l = lat + t * LAT + k;
      acc[t] = fmaf(w.x, l[0], fmaf(w.y, l[1], fmaf(w.z, l[2], fmaf(w.w, l[3], acc[t]))));
    }
  }
  const float base = (kk == 0) ? b0[h] : bh[(kk - 1) * HID + h];
#pragma unroll
  for (int t = 0; t < NT; ++t)
    beff[t * ((NHL + 1) * HID) + n] = acc[t] + base;
}

// ---------------------------------------------------------------------------
// Kernel 2: fused SIREN forward. One block = 128 pixels of one frame.
// Activations in LDS (fp16, padded pitch). Hidden layers via v_wmma_f32_16x16x32_f16.
// B fragments read from global in fragment-packed layout (L2-resident, coalesced).
// ---------------------------------------------------------------------------
__global__ __launch_bounds__(256)
void siren_forward_kernel(const float* __restrict__ coords,
                          const _Float16* __restrict__ weffh,
                          const float* __restrict__ weff0,
                          const float* __restrict__ beff,
                          const float* __restrict__ Wf,
                          const float* __restrict__ bf,
                          float* __restrict__ out) {
  __shared__ _Float16 act[TILE_P * PITCH];              // 67584 B

  const int t    = blockIdx.y;
  const int p0   = blockIdx.x * TILE_P;
  const int tid  = threadIdx.x;
  const int lane = tid & 31;
  const int wv   = tid >> 5;                            // 8 waves

  // ---- layer 0: CIN(3) -> HID(256), VALU ----
  {
    const int pl    = tid >> 1;                         // local pixel 0..127
    const int obase = (tid & 1) * 128;
    const float* cp = coords + (size_t)(t * HPIX + p0 + pl) * CIN;
    const float x = cp[0], y = cp[1], z = cp[2];
    const float* w0 = weff0 + (size_t)t * (HID * CIN);
    const float* bb = beff  + (size_t)t * ((NHL + 1) * HID);   // k = 0
#pragma unroll 4
    for (int o = obase; o < obase + 128; ++o) {
      float a = fmaf(w0[o * 3 + 0], x,
              fmaf(w0[o * 3 + 1], y,
              fmaf(w0[o * 3 + 2], z, bb[o])));
      act[pl * PITCH + o] = (_Float16)__sinf(OMEGA * a);
    }
  }
  __syncthreads();

  // ---- hidden layers: 3 x (256 -> 256) via WMMA ----
  const int n0     = wv * 32;                           // this wave's 32 output columns
  const int mrow   = lane & 15;
  const int khl    = (lane >> 4) * 8;                   // A layout: +8 halves for upper half-wave
  const int rowoff = (lane >> 4) * 8;                   // C/D: lanes 16-31 hold M+8

  for (int layer = 0; layer < NHL; ++layer) {
    const _Float16* __restrict__ wmat = weffh + ((size_t)(layer * NT + t) << 16);

    // prefetch next layer's weights for this frame (global_prefetch_b8)
    if (layer + 1 < NHL) {
      const _Float16* nxt = weffh + ((size_t)((layer + 1) * NT + t) << 16);
      __builtin_prefetch(nxt + ((size_t)tid << 8), 0, 1);   // 256 thr x 512B = 128KB
    }

    v8f zero = {};
    v8f acc[8][2];
#pragma unroll
    for (int r = 0; r < 8; ++r) { acc[r][0] = zero; acc[r][1] = zero; }

    for (int k0 = 0; k0 < HID; k0 += 32) {
      // B fragments: fragment-packed, one contiguous 1KB block per wave per fragment
      const _Float16* bp =
          wmat + (((size_t)(wv * 2) * 8 + (k0 >> 5)) << 9) + (lane << 4);
      U16H b0, b1;
      b0.h[0] = *(const v8h*)(bp);
      b0.h[1] = *(const v8h*)(bp + 8);
      b1.h[0] = *(const v8h*)(bp + 4096);                // next n_tile: +8 fragments
      b1.h[1] = *(const v8h*)(bp + 4096 + 8);

#pragma unroll
      for (int r = 0; r < 8; ++r) {
        U16H a;
        const _Float16* ap = &act[(r * 16 + mrow) * PITCH + k0 + khl];
        a.h[0] = *(const v8h*)(ap);
        a.h[1] = *(const v8h*)(ap + 16);
        acc[r][0] = __builtin_amdgcn_wmma_f32_16x16x32_f16(
            false, a.v, false, b0.v, (short)0, acc[r][0], false, false);
        acc[r][1] = __builtin_amdgcn_wmma_f32_16x16x32_f16(
            false, a.v, false, b1.v, (short)0, acc[r][1], false, false);
      }
    }
    __syncthreads();   // all waves finished reading act for this layer

    // epilogue: bias + sin(OMEGA*x) -> act (fp16)
    const float* bl = beff + ((size_t)t * (NHL + 1) + (layer + 1)) * HID;
    const float bias0 = bl[n0 + mrow];
    const float bias1 = bl[n0 + 16 + mrow];
#pragma unroll
    for (int r = 0; r < 8; ++r) {
#pragma unroll
      for (int e = 0; e < 8; ++e) {
        const int row = r * 16 + rowoff + e;
        act[row * PITCH + n0 + mrow] =
            (_Float16)__sinf(OMEGA * (acc[r][0][e] + bias0));
        act[row * PITCH + n0 + 16 + mrow] =
            (_Float16)__sinf(OMEGA * (acc[r][1][e] + bias1));
      }
    }
    __syncthreads();
  }

  // ---- final layer: HID(256) -> OUT(3), VALU ----
  if (tid < TILE_P) {
    const int pl = tid;
    float a0 = bf[0], a1 = bf[1], a2 = bf[2];
    for (int k = 0; k < HID; k += 8) {
      v8h h8 = *(const v8h*)&act[pl * PITCH + k];
#pragma unroll
      for (int e = 0; e < 8; ++e) {
        const float h = (float)h8[e];
        a0 = fmaf(h, Wf[0 * HID + k + e], a0);
        a1 = fmaf(h, Wf[1 * HID + k + e], a1);
        a2 = fmaf(h, Wf[2 * HID + k + e], a2);
      }
    }
    float* op = out + (size_t)(t * HPIX + p0 + pl) * NOUT;
    op[0] = a0; op[1] = a1; op[2] = a2;
  }
}

// ---------------------------------------------------------------------------
extern "C" void kernel_launch(void* const* d_in, const int* in_sizes, int n_in,
                              void* d_out, int out_size, void* d_ws, size_t ws_size,
                              hipStream_t stream) {
  const float* coords  = (const float*)d_in[0];
  const float* latents = (const float*)d_in[1];
  const float* W0      = (const float*)d_in[2];
  const float* b0      = (const float*)d_in[3];
  const float* Wh      = (const float*)d_in[4];
  const float* bh      = (const float*)d_in[5];
  const float* Wf      = (const float*)d_in[6];
  const float* bf      = (const float*)d_in[7];
  const float* HW0     = (const float*)d_in[8];
  const float* HWh     = (const float*)d_in[9];
  const float* HB      = (const float*)d_in[10];

  char* ws = (char*)d_ws;
  _Float16* weffh = (_Float16*)ws;                            // 3*32*65536*2 = 12,582,912 B
  float*    weff0 = (float*)(ws + 12582912);                  // 32*768*4     =     98,304 B
  float*    beff  = (float*)(ws + 12582912 + 98304);          // 32*4*256*4   =    131,072 B

  hyper_weights_kernel<<<dim3(3 * 65536 / 256), dim3(256), 0, stream>>>(latents, Wh, HWh, weffh);
  hyper_w0_kernel   <<<dim3(3), dim3(256), 0, stream>>>(latents, W0, HW0, weff0);
  hyper_bias_kernel <<<dim3(4), dim3(256), 0, stream>>>(latents, b0, bh, HB, beff);

  siren_forward_kernel<<<dim3(HPIX / TILE_P, NT), dim3(256), 0, stream>>>(
      coords, weffh, weff0, beff, Wf, bf, (float*)d_out);
}
